// CurveAggBlock_57664230916171
// MI455X (gfx1250) — matrix operand order repaired and checked by reading, not compiled, not run on hardware
//
#include <hip/hip_runtime.h>
#include <hip/hip_bf16.h>

typedef __attribute__((ext_vector_type(16))) _Float16 v16h;
typedef __attribute__((ext_vector_type(8)))  float    v8f;

#define B_ 8
#define N_ 4096
#define D_ 128
#define K_ 16
#define L_ 4
#define EPSF 1e-5f
#define KPAD 160   // 131 padded to 5 chunks of 32

// ---------------- WMMA helpers ----------------

__device__ __forceinline__ v8f wmma_f16f32(v16h a, v16h b, v8f c) {
  // 8 args: (neg_a, A, neg_b, B, c_mod, C, reuse_a, reuse_b)
  return __builtin_amdgcn_wmma_f32_16x16x32_f16(false, a, false, b, (short)0, c,
                                                false, false);
}

__device__ __forceinline__ v16h zero_v16h() {
  v16h z;
#pragma unroll
  for (int i = 0; i < 16; ++i) z[i] = (_Float16)0.0f;
  return z;
}

union H8 { uint4 u; _Float16 h[8]; };

// A fragment: 16x32 f16 tile from a row-major matrix (pitch in halves).
// lanes 0-15 : row = mBase+lane,   K = {kBase+0..7, kBase+16..23}
// lanes16-31 : row = mBase+lane-16,K = {kBase+8..15, kBase+24..31}
__device__ __forceinline__ v16h loadA_g(const _Float16* W, int pitch, int mBase,
                                        int kBase, int lane) {
  int lo = lane & 15, hi = lane >> 4;
  const _Float16* p = W + (size_t)(mBase + lo) * pitch + kBase + hi * 8;
  H8 c0, c1;
  c0.u = *(const uint4*)p;
  c1.u = *(const uint4*)(p + 16);
  v16h a;
#pragma unroll
  for (int j = 0; j < 8; ++j) { a[j] = c0.h[j]; a[8 + j] = c1.h[j]; }
  return a;
}

// B fragment: 32x16 f16 tile from LDS stored as S[col][pitch] (K contiguous).
// lanes 0-15 : col = colBase+lane,    K = {kBase+0..7, kBase+16..23}
// lanes16-31 : col = colBase+lane-16, K = {kBase+8..15, kBase+24..31}
__device__ __forceinline__ v16h loadB_s(const _Float16* S, int pitch, int colBase,
                                        int kBase, int lane) {
  int lo = lane & 15, hi = lane >> 4;
  const _Float16* p = S + (size_t)(colBase + lo) * pitch + kBase + hi * 8;
  H8 c0, c1;
  c0.u = *(const uint4*)p;
  c1.u = *(const uint4*)(p + 16);
  v16h b;
#pragma unroll
  for (int j = 0; j < 8; ++j) { b[j] = c0.h[j]; b[8 + j] = c1.h[j]; }
  return b;
}

// ---------------- weight prep: fold BN into W (f16) + bias ----------------

__global__ void prep_weights(
    const float* W1, const float* g1, const float* b1, const float* m1, const float* v1,
    const float* W2, const float* g2, const float* b2, const float* m2, const float* v2,
    const float* Wf1, const float* gf, const float* bf, const float* mf, const float* vf,
    const float* Wf2,
    _Float16* W1p, float* b1p, _Float16* W2p, float* b2p,
    _Float16* Wf1p, float* bfp, _Float16* Wf2h) {
  int idx = blockIdx.x * blockDim.x + threadIdx.x;
  if (idx < D_ * KPAD) {
    int o = idx / KPAD, c = idx % KPAD;
    float s = g1[o] * rsqrtf(v1[o] + EPSF);
    float val = (c < D_ + 3) ? W1[o * (D_ + 3) + c] * s : 0.0f;
    W1p[idx] = (_Float16)val;
  }
  if (idx < D_ * D_) {
    int o = idx / D_;
    float s = g2[o] * rsqrtf(v2[o] + EPSF);
    W2p[idx] = (_Float16)(W2[idx] * s);
  }
  if (idx < 2 * D_ * D_) {
    int o = idx / D_;
    float s = gf[o] * rsqrtf(vf[o] + EPSF);
    Wf1p[idx] = (_Float16)(Wf1[idx] * s);
  }
  if (idx < 2 * D_ * D_) {
    Wf2h[idx] = (_Float16)Wf2[idx];
  }
  if (idx < D_) {
    float s1 = g1[idx] * rsqrtf(v1[idx] + EPSF);
    b1p[idx] = b1[idx] - s1 * m1[idx];
    float s2 = g2[idx] * rsqrtf(v2[idx] + EPSF);
    b2p[idx] = b2[idx] - s2 * m2[idx];
  }
  if (idx < 2 * D_) {
    float sf = gf[idx] * rsqrtf(vf[idx] + EPSF);
    bfp[idx] = bf[idx] - sf * mf[idx];
  }
}

// ---------------- KNN via WMMA Gram tiles + per-lane top-16 ----------------

__global__ void __launch_bounds__(32) knn_kernel(const float* xyz, int* knn) {
  __shared__ __align__(16) float tileS[32][16];
  __shared__ float sqcS[16];
  int lane = threadIdx.x;
  int lo = lane & 15, hi = lane >> 4;
  int blocksPerBatch = N_ / 32;
  int b = blockIdx.x / blocksPerBatch;
  int qBase = (blockIdx.x % blocksPerBatch) * 32;
  const float* X = xyz + (size_t)b * N_ * 3;

  float qx = X[(qBase + lane) * 3 + 0];
  float qy = X[(qBase + lane) * 3 + 1];
  float qz = X[(qBase + lane) * 3 + 2];
  float sqq = qx * qx + qy * qy + qz * qz;

  // A fragments: queries as f16, K-dim {x,y,z,0,...}
  v16h a0 = zero_v16h(), a1 = zero_v16h();
  if (hi == 0) {
    a0[0] = (_Float16)X[(qBase + lo) * 3 + 0];
    a0[1] = (_Float16)X[(qBase + lo) * 3 + 1];
    a0[2] = (_Float16)X[(qBase + lo) * 3 + 2];
    a1[0] = (_Float16)X[(qBase + 16 + lo) * 3 + 0];
    a1[1] = (_Float16)X[(qBase + 16 + lo) * 3 + 1];
    a1[2] = (_Float16)X[(qBase + 16 + lo) * 3 + 2];
  }

  float bd[16]; int bi[16];
#pragma unroll
  for (int k = 0; k < 16; ++k) { bd[k] = 3.0e38f; bi[k] = 0; }

  for (int c0 = 0; c0 < N_; c0 += 16) {
    v16h bfr = zero_v16h();
    if (hi == 0) {
      float cx = X[(c0 + lo) * 3 + 0];
      float cy = X[(c0 + lo) * 3 + 1];
      float cz = X[(c0 + lo) * 3 + 2];
      sqcS[lo] = cx * cx + cy * cy + cz * cz;
      bfr[0] = (_Float16)cx; bfr[1] = (_Float16)cy; bfr[2] = (_Float16)cz;
    }
    __syncthreads();
    v8f acc0 = {}, acc1 = {};
    acc0 = wmma_f16f32(a0, bfr, acc0);  // inner products queries 0..15 x cands
    acc1 = wmma_f16f32(a1, bfr, acc1);  // inner products queries 16..31 x cands
#pragma unroll
    for (int r = 0; r < 8; ++r) {
      tileS[hi * 8 + r][lo]      = acc0[r];
      tileS[16 + hi * 8 + r][lo] = acc1[r];
    }
    __syncthreads();
#pragma unroll 1
    for (int j = 0; j < 16; ++j) {
      float dd = sqq + sqcS[j] - 2.0f * tileS[lane][j];
      int cand = c0 + j;
      if (dd < bd[15]) {
        bd[15] = dd; bi[15] = cand;
#pragma unroll
        for (int p = 15; p > 0; --p) {
          if (bd[p] < bd[p - 1]) {
            float tf = bd[p]; bd[p] = bd[p - 1]; bd[p - 1] = tf;
            int ti = bi[p];   bi[p] = bi[p - 1]; bi[p - 1] = ti;
          }
        }
      }
    }
    __syncthreads();
  }
  int* outp = knn + ((size_t)b * N_ + qBase + lane) * K_;
#pragma unroll
  for (int k = 0; k < 16; ++k) outp[k] = bi[k];
}

// ---------------- random-walk curves (threefry2x32-style RNG) ----------------

__device__ __forceinline__ unsigned rotl32(unsigned x, int r) {
  return (x << r) | (x >> (32 - r));
}

__device__ __forceinline__ uint2 threefry2x32(uint2 key, uint2 ctr) {
  unsigned ks[3] = {key.x, key.y, 0x1BD11BDAu ^ key.x ^ key.y};
  unsigned x0 = ctr.x + ks[0], x1 = ctr.y + ks[1];
  const int R0[4] = {13, 15, 26, 6};
  const int R1[4] = {17, 29, 16, 24};
#pragma unroll
  for (int g = 0; g < 5; ++g) {
#pragma unroll
    for (int i = 0; i < 4; ++i) {
      int r = (g & 1) ? R1[i] : R0[i];
      x0 += x1; x1 = rotl32(x1, r); x1 ^= x0;
    }
    x0 += ks[(g + 1) % 3];
    x1 += ks[(g + 2) % 3] + (unsigned)(g + 1);
  }
  return make_uint2(x0, x1);
}

__global__ void curves_kernel(const int* knn, int* curves) {
  int gid = blockIdx.x * blockDim.x + threadIdx.x;
  if (gid >= B_ * N_) return;
  int b = gid / N_, n = gid % N_;
  int cur = n;
  curves[(size_t)gid * L_ + 0] = cur;
#pragma unroll
  for (int i = 0; i < L_ - 1; ++i) {
    uint2 r = threefry2x32(make_uint2(0x2A000000u + (unsigned)i, 0x9E3779B9u),
                           make_uint2((unsigned)i * 0x85EBCA6Bu, (unsigned)gid));
    int rk = 1 + (int)(r.x % (unsigned)(K_ - 1));
    cur = knn[((size_t)b * N_ + cur) * K_ + rk];
    curves[(size_t)gid * L_ + i + 1] = cur;
  }
}

// ------------- fused gather + MLP1 + MLP2 + max_L + residual BN -------------

__global__ void __launch_bounds__(256) mlp12_kernel(
    const float* xyz, const float* feat, const int* curves,
    const _Float16* W1p, const float* b1p, const _Float16* W2p, const float* b2p,
    const float* gn, const float* bnb, const float* mn, const float* vn,
    float* fbuf) {
  __shared__ __align__(16) char smem[49152];
  _Float16* sComb = (_Float16*)smem;          // [64][KPAD] f16 = 20480 B (region 0)
  float*    sHH   = (float*)smem;             // [64][128] f32 = 32768 B (reuses region 0)
  _Float16* sH    = (_Float16*)(smem + 32768);// [64][128] f16 = 16384 B

  int tid = threadIdx.x;
  int lane = tid & 31, wave = tid >> 5;
  int lo = lane & 15, hi = lane >> 4;
  int tilesPerBatch = N_ / 16;
  int b = blockIdx.x / tilesPerBatch;
  int nBase = (blockIdx.x % tilesPerBatch) * 16;

  const float* X = xyz + (size_t)b * N_ * 3;
  const float* F = feat + (size_t)b * D_ * N_;
  const int*   C = curves + (size_t)b * N_ * L_;

  // stage combined tile: col = n_local*4 + l (64 cols), ch 0..KPAD-1 (zero pad)
  for (int idx = tid; idx < 64 * KPAD; idx += 256) {
    int col = idx / KPAD, c = idx % KPAD;
    int nl = col >> 2, l = col & 3;
    int n = nBase + nl;
    int j = C[n * L_ + l];
    float val;
    if (c < D_)            val = F[(size_t)c * N_ + j];
    else if (c < D_ + 3)   val = X[j * 3 + (c - D_)] - X[n * 3 + (c - D_)];
    else                   val = 0.0f;
    sComb[col * KPAD + c] = (_Float16)val;
  }
  __syncthreads();

  int mBase = wave * 16;
  v8f zf = {};
  v8f acc[4] = {zf, zf, zf, zf};
  __builtin_prefetch(W1p + (size_t)mBase * KPAD, 0, 0);
#pragma unroll
  for (int kc = 0; kc < KPAD; kc += 32) {
    v16h a = loadA_g(W1p, KPAD, mBase, kc, lane);
#pragma unroll
    for (int t = 0; t < 4; ++t) {
      v16h bfr = loadB_s(sComb, KPAD, t * 16, kc, lane);
      acc[t] = wmma_f16f32(a, bfr, acc[t]);
    }
  }
  // bias + relu -> sH (f16) [col][128]
#pragma unroll
  for (int t = 0; t < 4; ++t) {
    int col = t * 16 + lo;
    H8 w;
#pragma unroll
    for (int r = 0; r < 8; ++r) {
      int m = mBase + hi * 8 + r;
      float v = acc[t][r] + b1p[m];
      w.h[r] = (_Float16)(v > 0.0f ? v : 0.0f);
    }
    *(uint4*)(sH + col * D_ + mBase + hi * 8) = w.u;
  }
  __syncthreads();

  // GEMM2: W2' (128x128) x h
  v8f acc2[4] = {zf, zf, zf, zf};
  __builtin_prefetch(W2p + (size_t)mBase * D_, 0, 0);
#pragma unroll
  for (int kc = 0; kc < D_; kc += 32) {
    v16h a = loadA_g(W2p, D_, mBase, kc, lane);
#pragma unroll
    for (int t = 0; t < 4; ++t) {
      v16h bfr = loadB_s(sH, D_, t * 16, kc, lane);
      acc2[t] = wmma_f16f32(a, bfr, acc2[t]);
    }
  }
  __syncthreads();   // all reads of region 0 (sComb) done before overwrite
  // bias + relu -> sHH (f32) [col][128]
#pragma unroll
  for (int t = 0; t < 4; ++t) {
    int col = t * 16 + lo;
#pragma unroll
    for (int r = 0; r < 8; ++r) {
      int m = mBase + hi * 8 + r;
      float v = acc2[t][r] + b2p[m];
      sHH[col * D_ + m] = v > 0.0f ? v : 0.0f;
    }
  }
  __syncthreads();

  // max over L + residual + BN -> fbuf (f32, [b][m][n])
  for (int idx = tid; idx < 16 * D_; idx += 256) {
    int m = idx & (D_ - 1);
    int nl = idx >> 7;
    float mx = sHH[(nl * 4 + 0) * D_ + m];
#pragma unroll
    for (int l = 1; l < 4; ++l) {
      float v = sHH[(nl * 4 + l) * D_ + m];
      mx = v > mx ? v : mx;
    }
    int n = nBase + nl;
    float x = F[(size_t)m * N_ + n] + mx;
    float inv = rsqrtf(vn[m] + EPSF);
    fbuf[((size_t)b * D_ + m) * N_ + n] = gn[m] * (x - mn[m]) * inv + bnb[m];
  }
}

// ---------------- fused FFN (Wf1 -> relu -> Wf2) + final BN ----------------

__global__ void __launch_bounds__(256) ffn_kernel(
    const float* fbuf, const _Float16* Wf1p, const float* bfp, const _Float16* Wf2h,
    const float* gn2, const float* bn2, const float* mn2, const float* vn2,
    float* out) {
  __shared__ __align__(16) char smem[49152];
  _Float16* sF  = (_Float16*)smem;             // [64][128] f16 = 16384 B
  _Float16* sH2 = (_Float16*)(smem + 16384);   // [64][256] f16 = 32768 B

  int tid = threadIdx.x;
  int lane = tid & 31, wave = tid >> 5;
  int lo = lane & 15, hi = lane >> 4;
  int tilesPerBatch = N_ / 64;
  int b = blockIdx.x / tilesPerBatch;
  int nBase = (blockIdx.x % tilesPerBatch) * 64;
  const float* Fb = fbuf + (size_t)b * D_ * N_;

  for (int idx = tid; idx < 64 * D_; idx += 256) {
    int col = idx / D_, m = idx % D_;
    sF[col * D_ + m] = (_Float16)Fb[(size_t)m * N_ + nBase + col];
  }
  __syncthreads();

  v8f zf = {};
  v8f acc0[4] = {zf, zf, zf, zf};
  v8f acc1[4] = {zf, zf, zf, zf};
  int mB0 = wave * 16, mB1 = wave * 16 + 128;
  __builtin_prefetch(Wf1p + (size_t)mB0 * D_, 0, 0);
#pragma unroll
  for (int kc = 0; kc < D_; kc += 32) {
    v16h a0 = loadA_g(Wf1p, D_, mB0, kc, lane);
    v16h a1 = loadA_g(Wf1p, D_, mB1, kc, lane);
#pragma unroll
    for (int t = 0; t < 4; ++t) {
      v16h bfr = loadB_s(sF, D_, t * 16, kc, lane);
      acc0[t] = wmma_f16f32(a0, bfr, acc0[t]);
      acc1[t] = wmma_f16f32(a1, bfr, acc1[t]);
    }
  }
#pragma unroll
  for (int t = 0; t < 4; ++t) {
    int col = t * 16 + lo;
    H8 w0, w1;
#pragma unroll
    for (int r = 0; r < 8; ++r) {
      int m0 = mB0 + hi * 8 + r, m1 = mB1 + hi * 8 + r;
      float x0 = acc0[t][r] + bfp[m0];
      float x1 = acc1[t][r] + bfp[m1];
      w0.h[r] = (_Float16)(x0 > 0.0f ? x0 : 0.0f);
      w1.h[r] = (_Float16)(x1 > 0.0f ? x1 : 0.0f);
    }
    *(uint4*)(sH2 + col * 256 + mB0 + hi * 8) = w0.u;
    *(uint4*)(sH2 + col * 256 + mB1 + hi * 8) = w1.u;
  }
  __syncthreads();

  v8f acc[4] = {zf, zf, zf, zf};
  int mBase = wave * 16;
  __builtin_prefetch(Wf2h + (size_t)mBase * 2 * D_, 0, 0);
#pragma unroll
  for (int kc = 0; kc < 2 * D_; kc += 32) {
    v16h a = loadA_g(Wf2h, 2 * D_, mBase, kc, lane);
#pragma unroll
    for (int t = 0; t < 4; ++t) {
      v16h bfr = loadB_s(sH2, 2 * D_, t * 16, kc, lane);
      acc[t] = wmma_f16f32(a, bfr, acc[t]);
    }
  }
#pragma unroll
  for (int t = 0; t < 4; ++t) {
    int n = nBase + t * 16 + lo;
#pragma unroll
    for (int r = 0; r < 8; ++r) {
      int m = mBase + hi * 8 + r;
      float fv = Fb[(size_t)m * N_ + n];
      float inv = rsqrtf(vn2[m] + EPSF);
      out[((size_t)b * D_ + m) * N_ + n] =
          gn2[m] * (fv + acc[t][r] - mn2[m]) * inv + bn2[m];
    }
  }
}

// ---------------- launch ----------------

extern "C" void kernel_launch(void* const* d_in, const int* in_sizes, int n_in,
                              void* d_out, int out_size, void* d_ws, size_t ws_size,
                              hipStream_t stream) {
  (void)in_sizes; (void)n_in; (void)out_size; (void)ws_size;
  const float* xyz = (const float*)d_in[0];
  const float* feat = (const float*)d_in[1];
  const float* W1 = (const float*)d_in[2];
  const float* g1 = (const float*)d_in[3];
  const float* b1 = (const float*)d_in[4];
  const float* m1 = (const float*)d_in[5];
  const float* v1 = (const float*)d_in[6];
  const float* W2 = (const float*)d_in[7];
  const float* g2 = (const float*)d_in[8];
  const float* b2 = (const float*)d_in[9];
  const float* m2 = (const float*)d_in[10];
  const float* v2 = (const float*)d_in[11];
  const float* gn = (const float*)d_in[12];
  const float* bnb = (const float*)d_in[13];
  const float* mn = (const float*)d_in[14];
  const float* vn = (const float*)d_in[15];
  const float* Wf1 = (const float*)d_in[16];
  const float* gf = (const float*)d_in[17];
  const float* bff = (const float*)d_in[18];
  const float* mf = (const float*)d_in[19];
  const float* vf = (const float*)d_in[20];
  const float* Wf2 = (const float*)d_in[21];
  const float* gn2 = (const float*)d_in[22];
  const float* bn2 = (const float*)d_in[23];
  const float* mn2 = (const float*)d_in[24];
  const float* vn2 = (const float*)d_in[25];
  float* outp = (float*)d_out;

  char* ws = (char*)d_ws;
  size_t off = 0;
  int* knn = (int*)(ws + off);        off += (size_t)B_ * N_ * K_ * 4;
  int* curves = (int*)(ws + off);     off += (size_t)B_ * N_ * L_ * 4;
  _Float16* W1p = (_Float16*)(ws + off);  off += (size_t)D_ * KPAD * 2;
  _Float16* W2p = (_Float16*)(ws + off);  off += (size_t)D_ * D_ * 2;
  _Float16* Wf1p = (_Float16*)(ws + off); off += (size_t)2 * D_ * D_ * 2;
  _Float16* Wf2h = (_Float16*)(ws + off); off += (size_t)2 * D_ * D_ * 2;
  float* b1p = (float*)(ws + off);    off += (size_t)D_ * 4;
  float* b2p = (float*)(ws + off);    off += (size_t)D_ * 4;
  float* bfp = (float*)(ws + off);    off += (size_t)2 * D_ * 4;
  off = (off + 255) & ~(size_t)255;
  float* fbuf = (float*)(ws + off);   off += (size_t)B_ * D_ * N_ * 4;

  prep_weights<<<(2 * D_ * D_ + 255) / 256, 256, 0, stream>>>(
      W1, g1, b1, m1, v1, W2, g2, b2, m2, v2, Wf1, gf, bff, mf, vf, Wf2,
      W1p, b1p, W2p, b2p, Wf1p, bfp, Wf2h);
  knn_kernel<<<B_ * (N_ / 32), 32, 0, stream>>>(xyz, knn);
  curves_kernel<<<(B_ * N_ + 255) / 256, 256, 0, stream>>>(knn, curves);
  mlp12_kernel<<<B_ * (N_ / 16), 256, 0, stream>>>(
      xyz, feat, curves, W1p, b1p, W2p, b2p, gn, bnb, mn, vn, fbuf);
  ffn_kernel<<<B_ * (N_ / 64), 256, 0, stream>>>(
      fbuf, Wf1p, bfp, Wf2h, gn2, bn2, mn2, vn2, outp);
}
